// GIN_53944789238579
// MI455X (gfx1250) — matrix operand order repaired and checked by reading, not compiled, not run on hardware
//
#include <hip/hip_runtime.h>
#include <hip/hip_bf16.h>

typedef __bf16 bf16_t;
typedef __attribute__((ext_vector_type(16))) __bf16 v16bf;
typedef __attribute__((ext_vector_type(8)))  __bf16 v8bf;
typedef __attribute__((ext_vector_type(2)))  __bf16 v2bf;
typedef __attribute__((ext_vector_type(8)))  float  v8f;
typedef __attribute__((ext_vector_type(4)))  int    v4i;

#define ZSTR 136   // halves per LDS A-tile row: 272B -> 16B-aligned rows

// ---- feature-guarded fast paths -------------------------------------------
#if defined(__has_builtin)
#if __has_builtin(__builtin_amdgcn_global_load_async_to_lds_b128)
#define ASYNC_LDS 1
#endif
#if __has_builtin(__builtin_amdgcn_cvt_pk_bf16_f32)
#define HAVE_PK 1
#endif
#endif
#ifndef ASYNC_LDS
#define ASYNC_LDS 0
#endif
#ifndef HAVE_PK
#define HAVE_PK 0
#endif

__device__ __forceinline__ bf16_t f2bf(float f) {
  unsigned u = __builtin_bit_cast(unsigned, f);
  unsigned r = (u + 0x7FFFu + ((u >> 16) & 1u)) >> 16;   // round-to-nearest-even
  return __builtin_bit_cast(bf16_t, (unsigned short)r);
}

__device__ __forceinline__ v2bf pk2bf(float a, float b) {
#if HAVE_PK
  return __builtin_amdgcn_cvt_pk_bf16_f32(a, b);
#else
  v2bf r; r[0] = f2bf(a); r[1] = f2bf(b); return r;
#endif
}

__device__ __forceinline__ void async_wait0() {
#if ASYNC_LDS
#if __has_builtin(__builtin_amdgcn_s_wait_asynccnt)
  __builtin_amdgcn_s_wait_asynccnt(0);
#else
  asm volatile("s_wait_asynccnt 0" ::: "memory");
#endif
#endif
}

// ---------------------------------------------------------------- utilities
__global__ void zero_f32(float* __restrict__ p, long long n) {
  long long i = (long long)blockIdx.x * blockDim.x + threadIdx.x;
  long long stride = (long long)gridDim.x * blockDim.x;
  for (; i < n; i += stride) p[i] = 0.f;
}

// one-time f32 -> bf16 conversion of the five 128x128 weight matrices
__global__ void cvt_w_bf16(const float* __restrict__ Wa, const float* __restrict__ Wb,
                           const float* __restrict__ Wc, const float* __restrict__ Wd,
                           const float* __restrict__ We, bf16_t* o) {
  int i = blockIdx.x * blockDim.x + threadIdx.x;
  if (i >= 128 * 128) return;
  o[i]             = f2bf(Wa[i]);
  o[i + 16384]     = f2bf(Wb[i]);
  o[i + 16384 * 2] = f2bf(Wc[i]);
  o[i + 16384 * 3] = f2bf(Wd[i]);
  o[i + 16384 * 4] = f2bf(We[i]);
}

// agg[dst[e]] += h[src[e]]  (one wave per edge, float4 per lane)
__global__ void scatter_add128(const float* __restrict__ h,
                               const int* __restrict__ src,
                               const int* __restrict__ dst,
                               float* agg, long long E) {
  long long t = (long long)blockIdx.x * blockDim.x + threadIdx.x;
  long long e = t >> 5;
  if (e >= E) return;
  int lane = (int)(t & 31);
  long long s = src[e];
  long long d = dst[e];
  const float4 v = *(const float4*)(h + s * 128 + lane * 4);
  float* o = agg + d * 128 + lane * 4;
  atomicAdd(o + 0, v.x); atomicAdd(o + 1, v.y);
  atomicAdd(o + 2, v.z); atomicAdd(o + 3, v.w);
}

// g[batch[n]] += h[n]
__global__ void pool_add128(const float* __restrict__ h,
                            const int* __restrict__ batch,
                            float* g, long long N) {
  long long t = (long long)blockIdx.x * blockDim.x + threadIdx.x;
  long long n = t >> 5;
  if (n >= N) return;
  int lane = (int)(t & 31);
  long long b = batch[n];
  const float4 v = *(const float4*)(h + n * 128 + lane * 4);
  float* o = g + b * 128 + lane * 4;
  atomicAdd(o + 0, v.x); atomicAdd(o + 1, v.y);
  atomicAdd(o + 2, v.z); atomicAdd(o + 3, v.w);
}

// stage one 128x128 bf16 weight matrix into LDS (32KB), 128 threads
__device__ __forceinline__ void stage_weight(const bf16_t* W, bf16_t* sW, int tid) {
#if ASYNC_LDS
#pragma unroll
  for (int i = 0; i < 16; ++i) {
    int idx = (tid + i * 128) * 8;   // 8 bf16 = 16B per thread per iteration
    __builtin_amdgcn_global_load_async_to_lds_b128(
        (v4i*)(W + idx), (v4i*)(sW + idx), 0, 0);
  }
  async_wait0();
#else
  const uint4* s = (const uint4*)W;
  uint4* d = (uint4*)sW;
  for (int i = tid; i < 128 * 128 / 8; i += 128) d[i] = s[i];
#endif
}

// ------------------------------------------------- 16x128 @ 128x128 via WMMA
// A tile in LDS (row-major bf16, stride ZSTR), weights in LDS (row-major bf16,
// stride 128). Produces acc[nt][r]: N = nt*16 + lane%16, M = r + (lane/16)*8.
__device__ __forceinline__ void wmma_gemm_128(const bf16_t* zbase,
                                              const bf16_t* sW,
                                              int lane, v8f* acc) {
  const int l16  = lane & 15;
  const int half = lane >> 4;
#pragma unroll
  for (int nt = 0; nt < 8; ++nt) acc[nt] = (v8f){};
#pragma unroll
  for (int kc = 0; kc < 4; ++kc) {
    // A fragment: lane holds row M=l16; K = kc*32 + half*8 + [0..7] and +16
    const bf16_t* zr = zbase + l16 * ZSTR + kc * 32 + half * 8;
    v8bf alo = *(const v8bf*)(zr);
    v8bf ahi = *(const v8bf*)(zr + 16);
    v16bf a;
#pragma unroll
    for (int i = 0; i < 8; ++i) { a[i] = alo[i]; a[i + 8] = ahi[i]; }
    // B fragments: lane holds column N = nt*16 + l16; K = kc*32 + half*8 + [0..7] and +16
    const bf16_t* wr = sW + (kc * 32 + half * 8) * 128 + l16;
#pragma unroll
    for (int nt = 0; nt < 8; ++nt) {
      v16bf b;
#pragma unroll
      for (int i = 0; i < 8; ++i) { b[i] = wr[i * 128 + nt * 16]; b[i + 8] = wr[(i + 16) * 128 + nt * 16]; }
      acc[nt] = __builtin_amdgcn_wmma_f32_16x16x32_bf16(
          false, a, false, b, (short)0, acc[nt], false, false);
    }
  }
}

// Fused: out = [relu]( relu((h[+agg]) @ Wa + ba) [@ Wb + bb] )
// Block = 128 threads = 4 waves; each wave owns one 16-row tile.
__global__ __launch_bounds__(128, 1)
void gin_mlp_wmma(const float* __restrict__ h, const float* agg,
                  const bf16_t* __restrict__ Wa, const float* __restrict__ ba,
                  const bf16_t* Wb, const float* bb,
                  float* out, int nrows, int do_second, int relu_out) {
  __shared__ bf16_t sW[128 * 128];                 // 32 KB, Wa then Wb
  __shared__ float  sBias[128];
  __shared__ bf16_t sZ[4][16 * ZSTR];              // per-wave A tile

  const int tid  = threadIdx.x;
  const int wv   = tid >> 5;
  const int lane = tid & 31;
  const int l16  = lane & 15;
  const int half = lane >> 4;

  // stage Wa + bias a
  stage_weight(Wa, sW, tid);
  if (tid < 128) sBias[tid] = ba[tid];

  // stage z = h (+ agg) for this wave's 16 rows
  const long long rowBase = ((long long)blockIdx.x * 4 + wv) * 16;
  {
    bf16_t* zw = &sZ[wv][0];
#pragma unroll
    for (int r = 0; r < 16; ++r) {
      long long row = rowBase + r;
      float4 v = make_float4(0.f, 0.f, 0.f, 0.f);
      if (row < (long long)nrows) {
        v = *(const float4*)(h + row * 128 + lane * 4);
        if (agg) {
          float4 av = *(const float4*)(agg + row * 128 + lane * 4);
          v.x += av.x; v.y += av.y; v.z += av.z; v.w += av.w;
        }
      }
      bf16_t* p = &zw[r * ZSTR + lane * 4];
      *(v2bf*)(p)     = pk2bf(v.x, v.y);
      *(v2bf*)(p + 2) = pk2bf(v.z, v.w);
    }
  }
  __syncthreads();

  v8f acc[8];
  wmma_gemm_128(&sZ[wv][0], sW, lane, acc);

  // bias + inner ReLU
#pragma unroll
  for (int nt = 0; nt < 8; ++nt) {
    float bv = sBias[nt * 16 + l16];
#pragma unroll
    for (int r = 0; r < 8; ++r) {
      float v = acc[nt][r] + bv;
      acc[nt][r] = v > 0.f ? v : 0.f;
    }
  }

  if (do_second) {
    // write z2 back into the A tile (row-major bf16)
#pragma unroll
    for (int nt = 0; nt < 8; ++nt) {
      int n = nt * 16 + l16;
#pragma unroll
      for (int r = 0; r < 8; r += 2) {
        v2bf t = pk2bf(acc[nt][r], acc[nt][r + 1]);
        int m = r + half * 8;
        sZ[wv][m * ZSTR + n]       = t[0];
        sZ[wv][(m + 1) * ZSTR + n] = t[1];
      }
    }
    __syncthreads();
    // restage Wb + bias b over Wa
    stage_weight(Wb, sW, tid);
    if (tid < 128) sBias[tid] = bb[tid];
    __syncthreads();

    wmma_gemm_128(&sZ[wv][0], sW, lane, acc);
#pragma unroll
    for (int nt = 0; nt < 8; ++nt) {
      float bv = sBias[nt * 16 + l16];
#pragma unroll
      for (int r = 0; r < 8; ++r) {
        float v = acc[nt][r] + bv;
        if (relu_out) v = v > 0.f ? v : 0.f;
        acc[nt][r] = v;
      }
    }
  }

  // store f32 result (lanes 0-15 and 16-31 each cover 64B contiguous runs)
#pragma unroll
  for (int nt = 0; nt < 8; ++nt) {
    int n = nt * 16 + l16;
#pragma unroll
    for (int r = 0; r < 8; ++r) {
      long long m = rowBase + r + half * 8;
      if (m < (long long)nrows) out[m * 128 + n] = acc[nt][r];
    }
  }
}

// ------------------------------------------------------ head: 128->10 + lsm
__global__ void head_logsoftmax(const float* __restrict__ g2,
                                const float* __restrict__ Wl2,
                                const float* __restrict__ bl2,
                                float* out, int G) {
  int g = blockIdx.x * blockDim.x + threadIdx.x;
  if (g >= G) return;
  float logit[10];
#pragma unroll
  for (int j = 0; j < 10; ++j) logit[j] = bl2[j];
  const float* hr = g2 + (long long)g * 128;
  for (int k = 0; k < 128; ++k) {
    float hv = hr[k];
#pragma unroll
    for (int j = 0; j < 10; ++j) logit[j] += hv * Wl2[k * 10 + j];
  }
  float m = logit[0];
#pragma unroll
  for (int j = 1; j < 10; ++j) m = fmaxf(m, logit[j]);
  float s = 0.f;
#pragma unroll
  for (int j = 0; j < 10; ++j) s += __expf(logit[j] - m);
  float ls = __logf(s) + m;
#pragma unroll
  for (int j = 0; j < 10; ++j) out[g * 10 + j] = logit[j] - ls;
}

// ---------------------------------------------------------------------------
extern "C" void kernel_launch(void* const* d_in, const int* in_sizes, int n_in,
                              void* d_out, int out_size, void* d_ws, size_t ws_size,
                              hipStream_t stream) {
  const float* x    = (const float*)d_in[0];
  const int*   ei   = (const int*)d_in[1];
  const int*   batch= (const int*)d_in[2];
  const float* W1a  = (const float*)d_in[3];
  const float* b1a  = (const float*)d_in[4];
  const float* W1b  = (const float*)d_in[5];
  const float* b1b  = (const float*)d_in[6];
  const float* W2a  = (const float*)d_in[7];
  const float* b2a  = (const float*)d_in[8];
  const float* W2b  = (const float*)d_in[9];
  const float* b2b  = (const float*)d_in[10];
  const float* Wl1  = (const float*)d_in[11];
  const float* bl1  = (const float*)d_in[12];
  const float* Wl2  = (const float*)d_in[13];
  const float* bl2  = (const float*)d_in[14];

  const long long N = in_sizes[0] / 128;
  const long long E = in_sizes[1] / 2;
  const int G = 1024;
  const int* src = ei;
  const int* dst = ei + E;

  float* bufA = (float*)d_ws;                 // agg / h2
  float* bufB = bufA + (size_t)N * 128;       // h1
  float* gbuf = bufB + (size_t)N * 128;       // pooled graph features
  float* g2   = gbuf + (size_t)G * 128;       // relu(g @ Wl1 + bl1)
  bf16_t* wcv = (bf16_t*)(g2 + (size_t)G * 128);  // 5 x 128x128 bf16 weights
  const bf16_t* W1a_h = wcv;
  const bf16_t* W1b_h = wcv + 16384;
  const bf16_t* W2a_h = wcv + 16384 * 2;
  const bf16_t* W2b_h = wcv + 16384 * 3;
  const bf16_t* Wl1_h = wcv + 16384 * 4;

  const long long NF = N * 128;
  dim3 blk(256);
  int sgrid   = (int)((E * 32 + 255) / 256);
  int pgrid   = (int)((N * 32 + 255) / 256);
  int mgrid_n = (int)((N + 63) / 64);
  int mgrid_g = (G + 63) / 64;

  // one-time bf16 weight conversion (tiny)
  cvt_w_bf16<<<64, blk, 0, stream>>>(W1a, W1b, W2a, W2b, Wl1, wcv);

  // ---- layer 1: agg1 = scatter(x); h1 = relu(mlp2(x + agg1))
  zero_f32<<<2048, blk, 0, stream>>>(bufA, NF);
  scatter_add128<<<sgrid, blk, 0, stream>>>(x, src, dst, bufA, E);
  gin_mlp_wmma<<<mgrid_n, dim3(128), 0, stream>>>(x, bufA, W1a_h, b1a, W1b_h, b1b,
                                                  bufB, (int)N, 1, 1);
  // ---- layer 2: agg2 = scatter(h1); h2 = relu(mlp2(h1 + agg2))
  zero_f32<<<2048, blk, 0, stream>>>(bufA, NF);
  scatter_add128<<<sgrid, blk, 0, stream>>>(bufB, src, dst, bufA, E);
  gin_mlp_wmma<<<mgrid_n, dim3(128), 0, stream>>>(bufB, bufA, W2a_h, b2a, W2b_h, b2b,
                                                  bufA, (int)N, 1, 1);  // row-safe alias
  // ---- pool + graph MLP head
  zero_f32<<<64, blk, 0, stream>>>(gbuf, (long long)G * 128);
  pool_add128<<<pgrid, blk, 0, stream>>>(bufA, batch, gbuf, N);
  gin_mlp_wmma<<<mgrid_g, dim3(128), 0, stream>>>(gbuf, nullptr, Wl1_h, bl1,
                                                  nullptr, nullptr, g2, G, 0, 1);
  head_logsoftmax<<<(G + 255) / 256, blk, 0, stream>>>(g2, Wl2, bl2,
                                                       (float*)d_out, G);
}
